// LTAM_3350074490977
// MI455X (gfx1250) — matrix operand order repaired and checked by reading, not compiled, not run on hardware
//
#include <hip/hip_runtime.h>
#include <hip/hip_bf16.h>

// ---------------------------------------------------------------------------
// LTAM forward for MI455X (gfx1250, wave32, WMMA).
// Shapes fixed by setup_inputs(): n=1, t=4, c=128, h=w=192, s=4.
//   h4=w4=48, feat_len=2048, feat_num=2304, K_gemm=384*9=3456.
// Kernel 1: per-token cosine correlation + argmax (wave per token).
// Kernel 2: fused gather-fold + implicit-GEMM 3x3 conv via
//           v_wmma_f32_16x16x32_f16. Operands are staged in LDS in
//           WMMA-fragment order so every fragment is 2x ds_load_b128.
// ---------------------------------------------------------------------------

typedef __attribute__((ext_vector_type(16))) _Float16 v16h;
typedef __attribute__((ext_vector_type(8)))  float    v8f;

#define H4   48
#define W4   48
#define NTOK 2304          // 48*48
#define FLEN 2048          // c*s*s
#define HW   36864         // 192*192
#define KTOT 3456          // 384*9
#define KC   128           // K chunk (4 wmma steps of 32), 27 chunks
#define NGRP (KC / 32)     // 32-wide K groups per chunk
#define ASTR (KC + 8)      // a_sh row stride (halfs): 272B -> bank stride 4
#define BNS  40            // b_sh per-n stride (halfs): 80B -> bank stride 20
#define BGRP (128 * BNS)   // b_sh per-K-group stride (halfs)

__device__ __forceinline__ float wave_sum32(float v) {
  #pragma unroll
  for (int off = 16; off > 0; off >>= 1) v += __shfl_xor(v, off, 32);
  return v;
}

// swap bits 3 and 4 of k%32: fragment order {0..7,16..23,8..15,24..31}
__device__ __forceinline__ int afrag_pos(int km) {
  return (km & 7) | ((km & 8) << 1) | ((km & 16) >> 1);
}

// ---------------- Kernel 1: correlation / argmax / warp table --------------
// grid: 576 blocks x 128 threads (4 waves); wave w handles token p.
__global__ __launch_bounds__(128)
void ltam_corr_kernel(const float* __restrict__ curr_feat,      // (128,192,192)
                      const float* __restrict__ index_feat,     // (4,2048,48,48)
                      const float* __restrict__ location_feat,  // (8,48,48)
                      float* __restrict__ corr_soft,            // [2304]
                      int*   __restrict__ corr_index,           // [2304]
                      int*   __restrict__ wtab)                 // [4*2304]
{
  const int wave = threadIdx.x >> 5;
  const int lane = threadIdx.x & 31;
  const int p = blockIdx.x * 4 + wave;   // 576*4 == 2304, always in range
  const int y4 = p / W4;
  const int x4 = p - y4 * W4;

  // nearest-warp source index (the grid-sample math collapses to rint(loc))
  int ival[4];
  #pragma unroll
  for (int t = 0; t < 4; ++t) {
    float lx = location_feat[(2 * t + 0) * NTOK + p];
    float ly = location_feat[(2 * t + 1) * NTOK + p];
    int ix = (int)rintf(lx);   // round-half-to-even, matches jnp.round
    int iy = (int)rintf(ly);
    bool valid = (ix >= 0) & (ix < W4) & (iy >= 0) & (iy < H4);
    ival[t] = valid ? (iy * W4 + ix) : -1;
  }

  float cf2 = 0.f;
  float dot[4] = {0.f, 0.f, 0.f, 0.f};
  float qq [4] = {0.f, 0.f, 0.f, 0.f};

  // token = unfold(curr_feat): f = cidx*16 + kh*4 + kw
  for (int f = lane; f < FLEN; f += 32) {
    int cidx = f >> 4;
    int kh = (f >> 2) & 3;
    int kw = f & 3;
    float cf = curr_feat[cidx * HW + (y4 * 4 + kh) * 192 + (x4 * 4 + kw)];
    cf2 += cf * cf;
    #pragma unroll
    for (int t = 0; t < 4; ++t) {
      if (ival[t] >= 0) {
        float v = index_feat[(t * FLEN + f) * NTOK + ival[t]];
        qq[t]  += v * v;
        dot[t] += v * cf;
      }
    }
  }

  cf2 = wave_sum32(cf2);
  #pragma unroll
  for (int t = 0; t < 4; ++t) {
    qq[t]  = wave_sum32(qq[t]);
    dot[t] = wave_sum32(dot[t]);
  }

  if (lane == 0) {
    float ncf = fmaxf(sqrtf(cf2), 1e-12f);
    float best = -__builtin_inff();
    int bidx = 0;
    #pragma unroll
    for (int t = 0; t < 4; ++t) {
      float corr = dot[t] / (ncf * fmaxf(sqrtf(qq[t]), 1e-12f));
      if (corr > best) { best = corr; bidx = t; }   // first-max wins (argmax)
      wtab[t * NTOK + p] = ival[t];
    }
    corr_soft[p]  = best;
    corr_index[p] = bidx;
  }
}

// ---------------- Kernel 2: fused gather + WMMA conv + epilogue ------------
// grid: 2304 blocks (one per 16-pixel row tile) x 256 threads (8 waves).
// GEMM: D[m=pixel, n=cout] = sum_k A[m,k]*B[k,n], k = cin*9 + (ky*3+kx).
__global__ __launch_bounds__(256)
void ltam_conv_kernel(const float* __restrict__ s1,
                      const float* __restrict__ s2,
                      const float* __restrict__ s3,       // (4,2048,48,48) each
                      const float* __restrict__ fusion_w, // (128,384,3,3)
                      const float* __restrict__ fusion_b, // (128,)
                      const float* __restrict__ anchor,   // (128,192,192)
                      const float* __restrict__ corr_soft,
                      const int*   __restrict__ corr_index,
                      const int*   __restrict__ wtab,
                      float* __restrict__ out)            // (128,192,192)
{
  // fragment-order staging buffers (total ~45 KB)
  __shared__ _Float16 a_sh[16 * ASTR];          //  4,352 B
  __shared__ _Float16 b_sh[NGRP * BGRP];        // 40,960 B

  const int tile = blockIdx.x;           // 12 tiles per row, 192 rows
  const int y  = tile / 12;
  const int x0 = (tile - y * 12) * 16;

  const int tid  = threadIdx.x;
  const int wv   = tid >> 5;             // 0..7 -> cout slice
  const int lane = tid & 31;
  const int hh   = lane >> 4;            // lane half
  const int l    = lane & 15;

  const float* const sets[3] = {s1, s2, s3};

  v8f acc = {0.f, 0.f, 0.f, 0.f, 0.f, 0.f, 0.f, 0.f};

  for (int kb = 0; kb < KTOT; kb += KC) {
    __syncthreads();   // previous chunk's LDS reads done

    // prefetch next chunk's weight slab into cache (global_prefetch_b8)
    if (kb + KC < KTOT) {
      #pragma unroll
      for (int jj = 0; jj < 2; ++jj) {
        int e = tid + jj * 256;          // 512 cachelines of next B slab
        int co = e >> 2;
        int ln = e & 3;
        __builtin_prefetch(&fusion_w[co * KTOT + kb + KC + ln * 32], 0, 3);
      }
    }

    // ---- stage B chunk in fragment order: [k32grp][cout][k%32] ----
    #pragma unroll 4
    for (int j = 0; j < 64; ++j) {
      int e  = tid + j * 256;            // 0 .. 16383
      int co = e >> 7;
      int kl = e & (KC - 1);
      b_sh[(kl >> 5) * BGRP + co * BNS + afrag_pos(kl & 31)] =
          (_Float16)fusion_w[co * KTOT + kb + kl];
    }

    // ---- stage A chunk (fused gather-fold im2col) in fragment order ----
    #pragma unroll 4
    for (int j = 0; j < 8; ++j) {
      int e = tid + j * 256;             // 0 .. 2047
      int m  = e >> 7;
      int kl = e & (KC - 1);
      int kg = kb + kl;
      int cinp = kg / 9;
      int tap  = kg - cinp * 9;
      int ky = tap / 3;
      int kx = tap - ky * 3;
      int xs = x0 + m + kx - 1;
      int ys = y + ky - 1;
      float v = 0.f;
      if ((unsigned)xs < 192u && (unsigned)ys < 192u) {
        int p  = (ys >> 2) * W4 + (xs >> 2);
        int t  = corr_index[p];
        int iv = wtab[t * NTOK + p];
        if (iv >= 0) {
          int cidx = cinp & 127;
          int f = (cidx << 4) + ((ys & 3) << 2) + (xs & 3);
          v = sets[cinp >> 7][(t * FLEN + f) * NTOK + iv];
        }
      }
      a_sh[m * ASTR + (kl & ~31) + afrag_pos(kl & 31)] = (_Float16)v;
    }

    __syncthreads();

    // ---- 4 WMMA steps; each fragment is one 32B contiguous LDS read ----
    #pragma unroll
    for (int g = 0; g < NGRP; ++g) {
      v16h af = *(const v16h*)&a_sh[l * ASTR + g * 32 + 16 * hh];
      v16h bf = *(const v16h*)&b_sh[g * BGRP + (wv * 16 + l) * BNS + 16 * hh];
      acc = __builtin_amdgcn_wmma_f32_16x16x32_f16(
          /*neg_a=*/false, af, /*neg_b=*/false, bf,
          /*c_mod=*/(short)0, acc, /*reuse_a=*/false, /*reuse_b=*/false);
    }
  }

  // ---- epilogue: (conv + bias) * corr_soft_map + anchor ----
  const int cout = wv * 16 + l;          // D: N = lane%16
  const float bias = fusion_b[cout];
  const int prow = (y >> 2) * W4;
  #pragma unroll
  for (int r = 0; r < 8; ++r) {
    int m  = r + 8 * hh;                 // D: VGPR r -> M = r + 8*(lane/16)
    int xo = x0 + m;
    int pix = y * 192 + xo;
    float cs = corr_soft[prow + (xo >> 2)];
    out[cout * HW + pix] = (acc[r] + bias) * cs + anchor[cout * HW + pix];
  }
}

// ---------------------------------------------------------------------------
extern "C" void kernel_launch(void* const* d_in, const int* in_sizes, int n_in,
                              void* d_out, int out_size, void* d_ws, size_t ws_size,
                              hipStream_t stream) {
  const float* curr_feat  = (const float*)d_in[0];
  const float* index_feat = (const float*)d_in[1];
  const float* anchor     = (const float*)d_in[2];
  const float* s1         = (const float*)d_in[3];
  const float* s2         = (const float*)d_in[4];
  const float* s3         = (const float*)d_in[5];
  const float* loc        = (const float*)d_in[6];
  const float* fusion_w   = (const float*)d_in[7];
  const float* fusion_b   = (const float*)d_in[8];
  float* out = (float*)d_out;

  // workspace layout (~55 KB): corr_soft | corr_index | wtab
  float* corr_soft  = (float*)d_ws;
  int*   corr_index = (int*)(corr_soft + NTOK);
  int*   wtab       = corr_index + NTOK;

  ltam_corr_kernel<<<dim3(NTOK / 4), dim3(128), 0, stream>>>(
      curr_feat, index_feat, loc, corr_soft, corr_index, wtab);

  ltam_conv_kernel<<<dim3(NTOK), dim3(256), 0, stream>>>(
      s1, s2, s3, fusion_w, fusion_b, anchor, corr_soft, corr_index, wtab, out);
}
